// YOWO_36103495090532
// MI455X (gfx1250) — matrix operand order repaired and checked by reading, not compiled, not run on hardware
//
#include <hip/hip_runtime.h>
#include <stdint.h>

typedef __attribute__((ext_vector_type(16))) __bf16 v16bf;
typedef __attribute__((ext_vector_type(8)))  __bf16 v8bf;
typedef __attribute__((ext_vector_type(8)))  float  v8f;
typedef __attribute__((ext_vector_type(4)))  unsigned int u32x4;
typedef __attribute__((ext_vector_type(8)))  int i32x8;
typedef __attribute__((ext_vector_type(4)))  int i32x4;

#define NB    16
#define CIN   3072
#define HEAD  512
#define HW    1024
#define MPAD  160
#define MREAL 145
#define NANC  5
#define NCLS  24
#define MTOT  (HW*NANC)   // 5120
#define NTOP  1000
#define KT    64          // K-tile (2 WMMA k-steps per LDS tile)

// ---------------- CDNA5 Tensor Data Mover: 2D tile -> LDS ----------------
// D# built per CDNA5 ISA:
//  group0: [1:0]count=1, [63:32]lds_addr, [120:64]global_addr, [127:126]type=2
//  group1: [17:16]data_size=1(2B), [79:48]tensor_dim0, [111:80]tensor_dim1,
//          [127:112]tile_dim0, [143:128]tile_dim1, [207:160]tensor_dim0_stride
__device__ __forceinline__ void tdm_load_2d(uint32_t lds_off, const void* gptr,
                                            uint32_t tile_x_elems, uint32_t tile_rows,
                                            uint32_t stride_elems) {
  const uint32_t TD1 = 0x00100000u;  // generous tensor_dim1 (rows) for OOB check
  uint64_t ga = (uint64_t)(size_t)gptr;
  u32x4 g0;
  g0[0] = 1u;                                              // count=1 (valid user D#)
  g0[1] = lds_off;                                         // LDS byte address
  g0[2] = (uint32_t)ga;                                    // global_addr[31:0]
  g0[3] = (uint32_t)((ga >> 32) & 0x01FFFFFFu) | 0x80000000u; // addr[56:32] | type=2
  i32x8 g1;
  g1[0] = 0x00010000;                                      // data_size=1 -> 2 bytes
  g1[1] = (int)((stride_elems & 0xFFFFu) << 16);           // tensor_dim0[15:0]
  g1[2] = (int)(((stride_elems >> 16) & 0xFFFFu) | ((TD1 & 0xFFFFu) << 16));
  g1[3] = (int)(((TD1 >> 16) & 0xFFFFu) | (tile_x_elems << 16)); // dim1 hi | tile_dim0
  g1[4] = (int)(tile_rows & 0xFFFFu);                      // tile_dim1 (tile_dim2=0)
  g1[5] = (int)stride_elems;                               // tensor_dim0_stride[31:0]
  g1[6] = 0;                                               // stride hi / dim1_stride
  g1[7] = 0;
  i32x4 z4 = {0, 0, 0, 0};
  i32x8 z8 = {0, 0, 0, 0, 0, 0, 0, 0};
  // 6-arg form (clang-23 / therock-10.0 headers): (g0, g1, g2, g3, extra, cpol)
  __builtin_amdgcn_tensor_load_to_lds(g0, g1, z4, z4, z8, 0);
}
__device__ __forceinline__ void wait_tensor0() {
  __builtin_amdgcn_s_wait_tensorcnt(0);
}
__device__ __forceinline__ v16bf join8(v8bf lo, v8bf hi) {
  return __builtin_shufflevector(lo, hi, 0,1,2,3,4,5,6,7,8,9,10,11,12,13,14,15);
}
__device__ __forceinline__ float sigmoidf_(float x) { return 1.f / (1.f + __expf(-x)); }

__device__ __constant__ float ANCW[5] = {32.f, 64.f, 128.f, 128.f, 256.f};
__device__ __constant__ float ANCH[5] = {32.f, 128.f, 64.f, 128.f, 256.f};

// ---------------- prep kernels ----------------
__global__ void k_cvt_encw(const float* __restrict__ w, __bf16* __restrict__ o) {
  int i = blockIdx.x * blockDim.x + threadIdx.x;
  if (i < HEAD * CIN) o[i] = (__bf16)w[i];
}

__global__ void k_prep_pred(const float* __restrict__ pw, const float* __restrict__ pb,
                            __bf16* __restrict__ wo, float* __restrict__ bo) {
  int t = blockIdx.x * blockDim.x + threadIdx.x;           // MPAD*HEAD
  if (t >= MPAD * HEAD) return;
  int m = t >> 9, c = t & 511;
  wo[t] = (__bf16)((m < MREAL) ? pw[m * HEAD + c] : 0.f);
  if (c == 0) bo[m] = (m < MREAL) ? pb[m] : 0.f;
}

// feat [b][3072][1024] f32  ->  featT [b][1024][3072] bf16 (LDS tile transpose)
__global__ __launch_bounds__(256) void k_transpose_feat(const float* __restrict__ feat,
                                                        __bf16* __restrict__ out) {
  __shared__ float tile[32][33];
  int b = blockIdx.z, c0 = blockIdx.y * 32, n0 = blockIdx.x * 32;
  int tx = threadIdx.x, ty = threadIdx.y;                  // 32 x 8
#pragma unroll
  for (int r = 0; r < 4; ++r) {
    int c = c0 + ty + r * 8;
    tile[ty + r * 8][tx] = feat[((size_t)b * CIN + c) * HW + n0 + tx];
  }
  __syncthreads();
#pragma unroll
  for (int r = 0; r < 4; ++r) {
    int n = n0 + ty + r * 8;
    out[((size_t)b * HW + n) * CIN + c0 + tx] = (__bf16)tile[tx][ty + r * 8];
  }
}

// ---------------- GEMM 1: H^T = lrelu(W x X + b), bf16 WMMA + TDM feed ----------------
// Wb [512][3072] bf16, Xt [B][1024][3072] bf16, Ht [B][1024][512] bf16
__global__ __launch_bounds__(256) void k_gemm_enc(const __bf16* __restrict__ Wb,
                                                  const __bf16* __restrict__ Xt,
                                                  const float*  __restrict__ bias,
                                                  __bf16* __restrict__ Ht) {
  __shared__ __align__(16) __bf16 sA[2][128 * KT];   // 16 KB x2
  __shared__ __align__(16) __bf16 sB[2][128 * KT];   // 16 KB x2
  const int tid = threadIdx.x;
  const int b  = blockIdx.z;
  const int m0 = blockIdx.y * 128;
  const int n0 = blockIdx.x * 128;
  const int lane = tid & 31, w = tid >> 5;
  const int wm = w & 3, wn = w >> 2;
  const int g = lane >> 4, ln = lane & 15;
  const bool issuer = (tid < 32);                    // wave 0 drives the TDM

  v8f acc[2][4];
#pragma unroll
  for (int i = 0; i < 2; ++i)
#pragma unroll
    for (int j = 0; j < 4; ++j) acc[i][j] = v8f{0.f,0.f,0.f,0.f,0.f,0.f,0.f,0.f};

  const __bf16* gA = Wb + (size_t)m0 * CIN;
  const __bf16* gB = Xt + ((size_t)b * HW + n0) * CIN;

  auto issue = [&](int buf, int kto) {
    int k0 = kto * KT;
    tdm_load_2d((uint32_t)(size_t)&sA[buf][0], gA + k0, KT, 128, CIN);
    tdm_load_2d((uint32_t)(size_t)&sB[buf][0], gB + k0, KT, 128, CIN);
  };

  if (issuer) { issue(0, 0); wait_tensor0(); }
  __syncthreads();

  const int NKO = CIN / KT;                          // 48
  for (int kto = 0; kto < NKO; ++kto) {
    int cur = kto & 1;
    if (kto + 1 < NKO && issuer) issue(cur ^ 1, kto + 1);

#pragma unroll
    for (int s = 0; s < 2; ++s) {                    // two K=32 WMMA steps per tile
      v16bf aF[2], bF[4];
#pragma unroll
      for (int i = 0; i < 2; ++i) {
        int row = wm * 32 + i * 16 + ln;
        v8bf lo = *(const v8bf*)&sA[cur][row * KT + s * 32 + g * 8];
        v8bf hi = *(const v8bf*)&sA[cur][row * KT + s * 32 + 16 + g * 8];
        aF[i] = join8(lo, hi);
      }
#pragma unroll
      for (int j = 0; j < 4; ++j) {
        int col = wn * 64 + j * 16 + ln;
        v8bf blo = *(const v8bf*)&sB[cur][col * KT + s * 32 + g * 8];
        v8bf bhi = *(const v8bf*)&sB[cur][col * KT + s * 32 + 16 + g * 8];
        bF[j] = join8(blo, bhi);
      }
#pragma unroll
      for (int i = 0; i < 2; ++i)
#pragma unroll
        for (int j = 0; j < 4; ++j)
          acc[i][j] = __builtin_amdgcn_wmma_f32_16x16x32_bf16(
              false, aF[i], false, bF[j], (short)0, acc[i][j], false, false);
    }

    if (kto + 1 < NKO) {
      if (issuer) wait_tensor0();
      __syncthreads();
    }
  }

  // epilogue: +bias, leaky_relu(0.1), store H^T bf16 (8 consecutive M per lane -> 16B store)
#pragma unroll
  for (int i = 0; i < 2; ++i) {
    int mg = m0 + wm * 32 + i * 16 + g * 8;
    float bb[8];
#pragma unroll
    for (int r = 0; r < 8; ++r) bb[r] = bias[mg + r];
#pragma unroll
    for (int j = 0; j < 4; ++j) {
      int ng = n0 + wn * 64 + j * 16 + ln;
      v8bf o;
#pragma unroll
      for (int r = 0; r < 8; ++r) {
        float v = acc[i][j][r] + bb[r];
        v = v > 0.f ? v : 0.1f * v;
        o[r] = (__bf16)v;
      }
      *(v8bf*)(Ht + ((size_t)b * HW + ng) * HEAD + mg) = o;
    }
  }
}

// ---------------- GEMM 2: P^T = Wp x H + bp (TDM feed) ----------------
// Wp [160][512] bf16, Ht [B][1024][512] bf16, Pt [B][1024][160] f32
__global__ __launch_bounds__(256) void k_gemm_pred(const __bf16* __restrict__ Wp,
                                                   const __bf16* __restrict__ Ht,
                                                   const float*  __restrict__ biasP,
                                                   float* __restrict__ Pt) {
  __shared__ __align__(16) __bf16 sA[2][MPAD * KT];  // 20 KB x2
  __shared__ __align__(16) __bf16 sB[2][128 * KT];   // 16 KB x2
  const int tid = threadIdx.x;
  const int b  = blockIdx.z;
  const int n0 = blockIdx.x * 128;
  const int lane = tid & 31, w = tid >> 5;           // wave w owns 16 columns
  const int g = lane >> 4, ln = lane & 15;
  const bool issuer = (tid < 32);

  v8f acc[10];
#pragma unroll
  for (int t = 0; t < 10; ++t) acc[t] = v8f{0.f,0.f,0.f,0.f,0.f,0.f,0.f,0.f};

  const __bf16* gB = Ht + ((size_t)b * HW + n0) * HEAD;

  auto issue = [&](int buf, int kto) {
    int k0 = kto * KT;
    tdm_load_2d((uint32_t)(size_t)&sA[buf][0], Wp + k0, KT, MPAD, HEAD);
    tdm_load_2d((uint32_t)(size_t)&sB[buf][0], gB + k0, KT, 128, HEAD);
  };

  if (issuer) { issue(0, 0); wait_tensor0(); }
  __syncthreads();

  const int NKO = HEAD / KT;                         // 8
  for (int kto = 0; kto < NKO; ++kto) {
    int cur = kto & 1;
    if (kto + 1 < NKO && issuer) issue(cur ^ 1, kto + 1);

#pragma unroll
    for (int s = 0; s < 2; ++s) {
      int col = w * 16 + ln;
      v8bf blo = *(const v8bf*)&sB[cur][col * KT + s * 32 + g * 8];
      v8bf bhi = *(const v8bf*)&sB[cur][col * KT + s * 32 + 16 + g * 8];
      v16bf bF = join8(blo, bhi);
#pragma unroll
      for (int t = 0; t < 10; ++t) {
        int row = t * 16 + ln;
        v8bf lo = *(const v8bf*)&sA[cur][row * KT + s * 32 + g * 8];
        v8bf hi = *(const v8bf*)&sA[cur][row * KT + s * 32 + 16 + g * 8];
        v16bf aF = join8(lo, hi);
        acc[t] = __builtin_amdgcn_wmma_f32_16x16x32_bf16(
            false, aF, false, bF, (short)0, acc[t], false, false);
      }
    }
    if (kto + 1 < NKO) {
      if (issuer) wait_tensor0();
      __syncthreads();
    }
  }

  int ng = n0 + w * 16 + ln;
  float* orow = Pt + ((size_t)b * HW + ng) * MPAD;
#pragma unroll
  for (int t = 0; t < 10; ++t) {
    int mg = t * 16 + g * 8;
    float4 o0, o1;
    o0.x = acc[t][0] + biasP[mg + 0]; o0.y = acc[t][1] + biasP[mg + 1];
    o0.z = acc[t][2] + biasP[mg + 2]; o0.w = acc[t][3] + biasP[mg + 3];
    o1.x = acc[t][4] + biasP[mg + 4]; o1.y = acc[t][5] + biasP[mg + 5];
    o1.z = acc[t][6] + biasP[mg + 6]; o1.w = acc[t][7] + biasP[mg + 7];
    *(float4*)(orow + mg)     = o0;
    *(float4*)(orow + mg + 4) = o1;
  }
}

// ---------------- head math: sigmoid(conf) * softmax(cls) -> max/argmax ----------------
__global__ void k_head(const float* __restrict__ Pt, float* __restrict__ scores,
                       int* __restrict__ labels, float* __restrict__ regs) {
  int t = blockIdx.x * blockDim.x + threadIdx.x;     // b*5120 + n*5 + a
  if (t >= NB * MTOT) return;
  int a = t % NANC, n = (t / NANC) % HW, b = t / MTOT;
  const float* p = Pt + ((size_t)b * HW + n) * MPAD;
  float s = sigmoidf_(p[a]);
  const float* cl = p + NANC + a * NCLS;
  float mx = cl[0]; int best = 0;
#pragma unroll
  for (int j = 1; j < NCLS; ++j)
    if (cl[j] > mx) { mx = cl[j]; best = j; }
  float sum = 0.f;
#pragma unroll
  for (int j = 0; j < NCLS; ++j) sum += __expf(cl[j] - mx);
  scores[t] = s / sum;                               // = s * softmax at argmax
  labels[t] = best;
  const float* rg = p + NANC * (1 + NCLS) + a * 4;
  float* ro = regs + (size_t)t * 4;
  ro[0] = rg[0]; ro[1] = rg[1]; ro[2] = rg[2]; ro[3] = rg[3];
}

// ---------------- exact stable top-1000 per batch via rank counting ----------------
__global__ __launch_bounds__(1024) void k_topk(const float* __restrict__ scores,
                                               int* __restrict__ topIdx,
                                               float* __restrict__ topScore) {
  __shared__ float s[MTOT];
  int b = blockIdx.x, t = threadIdx.x;
  const float* sc = scores + (size_t)b * MTOT;
  for (int i = t; i < MTOT; i += 1024) s[i] = sc[i];
  __syncthreads();
  for (int i = t; i < MTOT; i += 1024) {
    float si = s[i];
    int rank = 0;
    for (int j = 0; j < MTOT; ++j) {
      float sj = s[j];
      rank += (sj > si) || (sj == si && j < i);      // stable descending order
    }
    if (rank < NTOP) { topIdx[b * NTOP + rank] = i; topScore[b * NTOP + rank] = si; }
  }
}

// ---------------- decode + sequential NMS + masked outputs ----------------
__global__ __launch_bounds__(1024) void k_nms(const int* __restrict__ topIdx,
                                              const float* __restrict__ topScore,
                                              const int* __restrict__ labels,
                                              const float* __restrict__ regs,
                                              float* __restrict__ out) {
  __shared__ float bx1[NTOP], by1[NTOP], bx2[NTOP], by2[NTOP], areaS[NTOP];
  __shared__ float obox[NTOP * 4], labS[NTOP], scrS[NTOP];
  __shared__ int keepS[NTOP];
  int b = blockIdx.x, t = threadIdx.x;

  if (t < NTOP) {
    int idx = topIdx[b * NTOP + t];
    float sc = topScore[b * NTOP + t];
    int lab = labels[(size_t)b * MTOT + idx];
    const float* rg = regs + ((size_t)b * MTOT + idx) * 4;
    int n = idx / NANC, a = idx % NANC;
    float cx = (float)(n & 31) * 16.f, cy = (float)(n >> 5) * 16.f;
    float x = sigmoidf_(rg[0]) * 16.f + cx;
    float y = sigmoidf_(rg[1]) * 16.f + cy;
    float ww = __expf(rg[2]) * ANCW[a];
    float hh = __expf(rg[3]) * ANCH[a];
    float x1 = fminf(fmaxf((x - 0.5f * ww) * (1.f / 512.f), 0.f), 1.f);
    float y1 = fminf(fmaxf((y - 0.5f * hh) * (1.f / 512.f), 0.f), 1.f);
    float x2 = fminf(fmaxf((x + 0.5f * ww) * (1.f / 512.f), 0.f), 1.f);
    float y2 = fminf(fmaxf((y + 0.5f * hh) * (1.f / 512.f), 0.f), 1.f);
    obox[t * 4 + 0] = x1; obox[t * 4 + 1] = y1; obox[t * 4 + 2] = x2; obox[t * 4 + 3] = y2;
    float off = (float)lab * 2.f;
    bx1[t] = x1 + off; by1[t] = y1 + off; bx2[t] = x2 + off; by2[t] = y2 + off;
    areaS[t] = (x2 - x1) * (y2 - y1);
    keepS[t] = (sc >= 0.05f) ? 1 : 0;
    labS[t] = (float)lab; scrS[t] = sc;
  }
  __syncthreads();

  for (int i = 0; i < NTOP; ++i) {
    if (keepS[i]) {                                  // uniform read across block
      if (t < NTOP && t > i) {
        float w = fmaxf(1e-28f, fminf(bx2[t], bx2[i]) - fmaxf(bx1[t], bx1[i]));
        float h = fmaxf(1e-28f, fminf(by2[t], by2[i]) - fmaxf(by1[t], by1[i]));
        float inter = w * h;
        float iou = inter / (areaS[t] + areaS[i] - inter + 1e-14f);
        if (iou > 0.6f) keepS[t] = 0;
      }
    }
    __syncthreads();
  }

  if (t < NTOP) {
    float m = keepS[t] ? 1.f : 0.f;
    out[b * NTOP + t] = scrS[t] * m;                           // scores * mask
    out[NB * NTOP + b * NTOP + t] = labS[t];                   // labels (unmasked)
    float* ob = out + 2 * NB * NTOP + ((size_t)b * NTOP + t) * 4;
    ob[0] = obox[t * 4 + 0] * m; ob[1] = obox[t * 4 + 1] * m;
    ob[2] = obox[t * 4 + 2] * m; ob[3] = obox[t * 4 + 3] * m;
    out[6 * NB * NTOP + b * NTOP + t] = m;                     // keep mask
  }
}

// ---------------- launcher ----------------
extern "C" void kernel_launch(void* const* d_in, const int* in_sizes, int n_in,
                              void* d_out, int out_size, void* d_ws, size_t ws_size,
                              hipStream_t stream) {
  const float* feat   = (const float*)d_in[0];  // [16][3072][32][32]
  const float* enc_w  = (const float*)d_in[1];  // [512][3072]
  const float* enc_b  = (const float*)d_in[2];  // [512]
  const float* pred_w = (const float*)d_in[3];  // [145][512]
  const float* pred_b = (const float*)d_in[4];  // [145]
  float* out = (float*)d_out;

  char* base = (char*)d_ws;
  size_t off = 0;
  auto alloc = [&](size_t bytes) -> char* {
    char* p = base + off;
    off = (off + bytes + 255) & ~(size_t)255;
    return p;
  };
  __bf16* featT   = (__bf16*)alloc((size_t)NB * HW * CIN * 2);     // ~100.7 MB
  __bf16* encwB   = (__bf16*)alloc((size_t)HEAD * CIN * 2);        // 3 MB
  __bf16* predwB  = (__bf16*)alloc((size_t)MPAD * HEAD * 2);
  float*  predbP  = (float*)alloc(MPAD * 4);
  __bf16* hT      = (__bf16*)alloc((size_t)NB * HW * HEAD * 2);    // 16.8 MB
  float*  Pt      = (float*)alloc((size_t)NB * HW * MPAD * 4);     // 10.5 MB
  float*  scoresA = (float*)alloc((size_t)NB * MTOT * 4);
  int*    labelsA = (int*)alloc((size_t)NB * MTOT * 4);
  float*  regsA   = (float*)alloc((size_t)NB * MTOT * 16);
  int*    topIdx  = (int*)alloc((size_t)NB * NTOP * 4);
  float*  topScr  = (float*)alloc((size_t)NB * NTOP * 4);
  (void)ws_size; (void)in_sizes; (void)n_in; (void)out_size;

  k_cvt_encw<<<(HEAD * CIN + 255) / 256, 256, 0, stream>>>(enc_w, encwB);
  k_prep_pred<<<(MPAD * HEAD + 255) / 256, 256, 0, stream>>>(pred_w, pred_b, predwB, predbP);
  k_transpose_feat<<<dim3(HW / 32, CIN / 32, NB), dim3(32, 8), 0, stream>>>(feat, featT);
  k_gemm_enc<<<dim3(HW / 128, HEAD / 128, NB), 256, 0, stream>>>(encwB, featT, enc_b, hT);
  k_gemm_pred<<<dim3(HW / 128, 1, NB), 256, 0, stream>>>(predwB, hT, predbP, Pt);
  k_head<<<(NB * MTOT + 255) / 256, 256, 0, stream>>>(Pt, scoresA, labelsA, regsA);
  k_topk<<<NB, 1024, 0, stream>>>(scoresA, topIdx, topScr);
  k_nms<<<NB, 1024, 0, stream>>>(topIdx, topScr, labelsA, regsA, out);
}
